// MoE_17789754540397
// MI455X (gfx1250) — matrix operand order repaired and checked by reading, not compiled, not run on hardware
//
#include <hip/hip_runtime.h>

// Dims from the reference
#define BB 8
#define SS 512
#define VV 16384
#define DD 512
#define EE 4

typedef __attribute__((ext_vector_type(16))) __bf16 v16bf;
typedef __attribute__((ext_vector_type(2)))  __bf16 v2bf;
typedef __attribute__((ext_vector_type(8)))  float  v8f;

// ---------- helpers ----------
__device__ __forceinline__ unsigned int pkbf2(float a, float b) {
    // native f32->bf16 (lowers to v_cvt_pk_bf16_f32 when available)
    v2bf v = { (__bf16)a, (__bf16)b };
    return __builtin_bit_cast(unsigned int, v);
}

__device__ __forceinline__ v16bf ld_frag(const unsigned short* p, int off2) {
    // two 16B LDS loads -> one 16-element bf16 fragment
    struct Pair { uint4 a, b; } q;
    q.a = *(const uint4*)(p);
    q.b = *(const uint4*)(p + off2);
    return __builtin_bit_cast(v16bf, q);
}

// ---------- kernel 1: router (one wave per batch) ----------
__global__ __launch_bounds__(256) void router_kernel(
    const int* __restrict__ x, const float* __restrict__ Wr, const float* __restrict__ br,
    int* __restrict__ experts, float* __restrict__ weights)
{
    const int b    = threadIdx.x >> 5;
    const int lane = threadIdx.x & 31;
    if (b >= BB) return;

    float p[EE] = {0.f, 0.f, 0.f, 0.f};
    for (int s = lane; s < SS; s += 32) {
        const float xv = (float)x[b * SS + s];
        #pragma unroll
        for (int e = 0; e < EE; ++e) p[e] += xv * Wr[s * EE + e];
    }
    #pragma unroll
    for (int off = 16; off > 0; off >>= 1) {
        #pragma unroll
        for (int e = 0; e < EE; ++e) p[e] += __shfl_down(p[e], off, 32);
    }
    if (lane == 0) {
        #pragma unroll
        for (int e = 0; e < EE; ++e) p[e] += br[e];
        float m = p[0];
        #pragma unroll
        for (int e = 1; e < EE; ++e) m = fmaxf(m, p[e]);
        float pr[EE], se = 0.f;
        #pragma unroll
        for (int e = 0; e < EE; ++e) { pr[e] = __expf(p[e] - m); se += pr[e]; }
        #pragma unroll
        for (int e = 0; e < EE; ++e) pr[e] /= se;
        // stable top-2 (earlier index wins ties, like lax.top_k)
        float v0 = -1.f, v1 = -1.f; int i0 = 0, i1 = 0;
        #pragma unroll
        for (int e = 0; e < EE; ++e) {
            if (pr[e] > v0) { v1 = v0; i1 = i0; v0 = pr[e]; i0 = e; }
            else if (pr[e] > v1) { v1 = pr[e]; i1 = e; }
        }
        const float inv = 1.f / (v0 + v1);
        experts[2 * b + 0] = i0;  experts[2 * b + 1] = i1;
        weights[2 * b + 0] = v0 * inv;  weights[2 * b + 1] = v1 * inv;
    }
}

// ---------- kernel 2: base[i][v] and per-batch constant cvec[b][v] (f32) ----------
__global__ __launch_bounds__(256) void base_cvec_kernel(
    const float* __restrict__ emb, const float* __restrict__ wp,
    const int* __restrict__ experts, const float* __restrict__ weights,
    float* __restrict__ cvec)
{
    const int v = blockIdx.x * 256 + threadIdx.x;   // V/256 blocks
    float base[EE];
    #pragma unroll
    for (int i = 0; i < EE; ++i) {
        const float* er = emb + (size_t)i * VV * DD;       // row for token id 0
        const float* wc = wp  + (size_t)i * DD * VV + v;
        float s = 0.f;
        for (int d = 0; d < DD; ++d) s = fmaf(er[d], wc[(size_t)d * VV], s);
        base[i] = s;
    }
    const float tb = base[0] + base[1] + base[2] + base[3];
    #pragma unroll
    for (int b = 0; b < BB; ++b) {
        const float c = tb - weights[2 * b + 0] * base[experts[2 * b + 0]]
                           - weights[2 * b + 1] * base[experts[2 * b + 1]];
        cvec[(size_t)b * VV + v] = c;
    }
}

// ---------- kernel 3: main gathered GEMM, bf16 WMMA, f32 accumulate ----------
// grid: (V/128, S/128, B), block: 256 threads = 8 waves (2x4 wave tiling).
// Double-buffered: global loads for tile it+1 are issued before the barrier,
// conversion+LDS store happens after computing tile it (register pipeline).
__global__ __launch_bounds__(256) void moe_main_kernel(
    const int*   __restrict__ x,   const float* __restrict__ emb,
    const float* __restrict__ wp,  const int*   __restrict__ experts,
    const float* __restrict__ weights, const float* __restrict__ cvec,
    float* __restrict__ out)
{
    __shared__ __align__(16) unsigned short Asm[2][128 * 72];  // A tile, [m][k], bf16
    __shared__ __align__(16) unsigned short Bsm[2][128 * 72];  // B tile, transposed [n][k], bf16
    __shared__ int s_tok[128];

    const int b     = blockIdx.z;
    const int sTile = blockIdx.y * 128;
    const int vTile = blockIdx.x * 128;
    const int t     = threadIdx.x;
    const int lane  = t & 31;
    const int wave  = t >> 5;
    const int wm    = wave & 1;     // 64-row half of the 128x128 tile
    const int wn    = wave >> 1;    // 32-col quarter
    const int half  = lane >> 4;    // lane half for WMMA operand layout
    const int lrow  = lane & 15;

    if (t < 128) s_tok[t] = x[b * SS + sTile + t];
    __syncthreads();                // tokens visible before first gather

    // uniform per-slot state
    const int   e0 = experts[2 * b + 0], e1 = experts[2 * b + 1];
    const float w0 = weights[2 * b + 0], w1 = weights[2 * b + 1];
    const float* embE0 = emb + (size_t)e0 * VV * DD;
    const float* embE1 = emb + (size_t)e1 * VV * DD;
    const float* wpE0  = wp  + (size_t)e0 * DD * VV + vTile;
    const float* wpE1  = wp  + (size_t)e1 * DD * VV + vTile;

    v8f acc[4][2];
    #pragma unroll
    for (int i = 0; i < 4; ++i)
        #pragma unroll
        for (int j = 0; j < 2; ++j)
            acc[i][j] = (v8f){0.f, 0.f, 0.f, 0.f, 0.f, 0.f, 0.f, 0.f};

    const int bn  = t & 127;   // B-stage: column n handled by this thread
    const int bkh = t >> 7;    // B-stage: k-quad phase (0/1)

    float4 aReg[8];            // held A gather data (next tile)
    float4 bReg[8];            // held B data (next tile)
    float  wStage;

    // issue global loads for tile 'it' (16 tiles: slot = it>>3, kt = it&7)
    auto loadTile = [&](int it) {
        const int slot  = it >> 3;
        const int dBase = (it & 7) * 64;
        const float* embE = slot ? embE1 : embE0;
        const float* wpE  = slot ? wpE1  : wpE0;
        wStage = slot ? w1 : w0;
        #pragma unroll
        for (int q = 0; q < 8; ++q) {
            const int idx = t + q * 256;
            const int m = idx >> 4, c4 = idx & 15;
            aReg[q] = *(const float4*)(embE + (size_t)s_tok[m] * DD + dBase + c4 * 4);
        }
        #pragma unroll
        for (int q = 0; q < 8; ++q) {
            const int kq = bkh + q * 2;                     // k = 4*kq .. +3
            const float* pc = wpE + (size_t)(dBase + kq * 4) * VV + bn;
            bReg[q] = make_float4(pc[0], pc[(size_t)VV], pc[2 * (size_t)VV], pc[3 * (size_t)VV]);
        }
    };

    // convert held registers to bf16 and store into LDS buffer 'buf'
    auto storeTile = [&](int buf) {
        #pragma unroll
        for (int q = 0; q < 8; ++q) {
            const int idx = t + q * 256;
            const int m = idx >> 4, c4 = idx & 15;
            const unsigned lo = pkbf2(wStage * aReg[q].x, wStage * aReg[q].y);
            const unsigned hi = pkbf2(wStage * aReg[q].z, wStage * aReg[q].w);
            *(uint2*)&Asm[buf][m * 72 + c4 * 4] = make_uint2(lo, hi);
        }
        #pragma unroll
        for (int q = 0; q < 8; ++q) {
            const int kq = bkh + q * 2;
            const unsigned lo = pkbf2(bReg[q].x, bReg[q].y);
            const unsigned hi = pkbf2(bReg[q].z, bReg[q].w);
            *(uint2*)&Bsm[buf][bn * 72 + kq * 4] = make_uint2(lo, hi);
        }
    };

    // WMMA compute on LDS buffer 'buf': two K=32 steps, 8 WMMAs each per wave
    auto computeTile = [&](int buf) {
        #pragma unroll
        for (int k0 = 0; k0 < 64; k0 += 32) {
            v16bf af[4], bfr[2];
            #pragma unroll
            for (int i = 0; i < 4; ++i) {
                const int row = wm * 64 + i * 16 + lrow;
                af[i] = ld_frag(&Asm[buf][row * 72 + k0 + half * 8], 16);
            }
            #pragma unroll
            for (int j = 0; j < 2; ++j) {
                const int col = wn * 32 + j * 16 + lrow;
                bfr[j] = ld_frag(&Bsm[buf][col * 72 + k0 + half * 8], 16);
            }
            #pragma unroll
            for (int i = 0; i < 4; ++i)
                #pragma unroll
                for (int j = 0; j < 2; ++j)
                    acc[i][j] = __builtin_amdgcn_wmma_f32_16x16x32_bf16(
                        false, af[i], false, bfr[j], (short)0, acc[i][j], false, false);
        }
    };

    // ---- software pipeline over 16 K-tiles (2 slots x 8) ----
    loadTile(0);
    storeTile(0);
    for (int it = 0; it < 16; ++it) {
        const int cur = it & 1;
        if (it + 1 < 16) loadTile(it + 1);     // global latency hidden by compute below
        __syncthreads();                        // LDS stores for buf[cur] visible
        computeTile(cur);
        if (it + 1 < 16) storeTile(1 - cur);    // write other buffer; safe (barrier-ordered)
    }

    // --- epilogue: out = cvec + acc (C/D layout: VGPR r -> M = r + 8*half, N = lrow) ---
    float cv[2];
    cv[0] = cvec[(size_t)b * VV + vTile + wn * 32 + 0  + lrow];
    cv[1] = cvec[(size_t)b * VV + vTile + wn * 32 + 16 + lrow];
    #pragma unroll
    for (int i = 0; i < 4; ++i) {
        #pragma unroll
        for (int j = 0; j < 2; ++j) {
            const int gcol = vTile + wn * 32 + j * 16 + lrow;
            #pragma unroll
            for (int r = 0; r < 8; ++r) {
                const int grow = sTile + wm * 64 + i * 16 + r + half * 8;
                out[((size_t)b * SS + grow) * VV + gcol] = cv[j] + acc[i][j][r];
            }
        }
    }
}

// ---------- launch ----------
extern "C" void kernel_launch(void* const* d_in, const int* in_sizes, int n_in,
                              void* d_out, int out_size, void* d_ws, size_t ws_size,
                              hipStream_t stream) {
    const int*   x   = (const int*)  d_in[0];   // [B,S] int32
    const float* Wr  = (const float*)d_in[1];   // [S,E]
    const float* br  = (const float*)d_in[2];   // [E]
    const float* emb = (const float*)d_in[3];   // [E,V,D]
    const float* wp  = (const float*)d_in[4];   // [E,D,V]
    float* out = (float*)d_out;                 // [B,S,V]

    int*   experts = (int*)d_ws;                         // 2*B ints
    float* weights = (float*)((char*)d_ws + 64);         // 2*B floats
    float* cvec    = (float*)((char*)d_ws + 256);        // B*V floats (512 KB)

    router_kernel<<<1, 256, 0, stream>>>(x, Wr, br, experts, weights);
    base_cvec_kernel<<<VV / 256, 256, 0, stream>>>(emb, wp, experts, weights, cvec);

    dim3 grid(VV / 128, SS / 128, BB);
    moe_main_kernel<<<grid, 256, 0, stream>>>(x, emb, wp, experts, weights, cvec, out);
}
